// LSTM_13657996001437
// MI455X (gfx1250) — compile-verified, matrix operand order checked by reading, not run on hardware
//
#include <hip/hip_runtime.h>

typedef __attribute__((ext_vector_type(16))) __bf16 v16bf;
typedef __attribute__((ext_vector_type(8)))  float  v8f;

// Problem dims (fixed by the reference)
constexpr int B = 32, T = 2048, D = 1024, H = 1024, G = 4096; // G = 4H

// ---------------- fp32 -> bf16 conversion ----------------
__global__ void cvt_bf16(const float* __restrict__ src, __bf16* __restrict__ dst, long n) {
    long i = (long)blockIdx.x * blockDim.x + threadIdx.x;
    if (i < n) dst[i] = (__bf16)src[i];
}

// W [K,N] fp32  ->  WT [N,K] bf16   (so B-operand columns are contiguous)
__global__ void transpose_bf16(const float* __restrict__ W, __bf16* __restrict__ WT,
                               int K, int N) {
    long i = (long)blockIdx.x * blockDim.x + threadIdx.x;
    if (i >= (long)K * N) return;
    int n = (int)(i % N);
    int k = (int)(i / N);
    WT[(long)n * K + k] = (__bf16)W[(long)k * N + n];
}

// ---------------- big GEMM: xW = Xbf[M,K] * WiT[N,K]^T + bias ----------------
// One wave computes a 16x64 strip: 4 accumulators reuse one A fragment.
__global__ void __launch_bounds__(256) gemm_bf16(
    const __bf16* __restrict__ A,    // [M,K] row-major bf16
    const __bf16* __restrict__ BT_,  // [N,K] row-major bf16 (i.e. B transposed)
    const float*  __restrict__ bias, // [N]
    float*        __restrict__ Cmat, // [M,N] fp32
    int M, int K, int N) {
    int wave = (int)((blockIdx.x * (long)blockDim.x + threadIdx.x) >> 5);
    int lane = threadIdx.x & 31;
    int nGroups = N >> 6;            // strips of 64 columns
    int mTile = wave / nGroups;
    int nGrp  = wave % nGroups;
    if (mTile * 16 >= M) return;     // uniform across wave
    int l16 = lane & 15, half = lane >> 4;

    const __bf16* arow  = A + (long)(mTile * 16 + l16) * K + half * 16;
    long bbase = (long)(nGrp * 64 + l16) * K + half * 16;

    v8f acc0 = {}, acc1 = {}, acc2 = {}, acc3 = {};
    for (int k0 = 0; k0 < K; k0 += 32) {
        v16bf a  = *(const v16bf*)(arow + k0);
        v16bf b0 = *(const v16bf*)(BT_ + bbase + k0);
        v16bf b1 = *(const v16bf*)(BT_ + bbase + (long)16 * K + k0);
        v16bf b2 = *(const v16bf*)(BT_ + bbase + (long)32 * K + k0);
        v16bf b3 = *(const v16bf*)(BT_ + bbase + (long)48 * K + k0);
        __builtin_prefetch(arow + k0 + 64, 0, 3);   // near-cache prefetch
        acc0 = __builtin_amdgcn_wmma_f32_16x16x32_bf16(false, a, false, b0, (short)0, acc0, false, false);
        acc1 = __builtin_amdgcn_wmma_f32_16x16x32_bf16(false, a, false, b1, (short)0, acc1, false, false);
        acc2 = __builtin_amdgcn_wmma_f32_16x16x32_bf16(false, a, false, b2, (short)0, acc2, false, false);
        acc3 = __builtin_amdgcn_wmma_f32_16x16x32_bf16(false, a, false, b3, (short)0, acc3, false, false);
    }
    // C/D layout: VGPR r -> row (half*8 + r), column = lane&15
    int col0 = nGrp * 64 + l16;
    float bi0 = bias[col0], bi1 = bias[col0 + 16], bi2 = bias[col0 + 32], bi3 = bias[col0 + 48];
#pragma unroll
    for (int r = 0; r < 8; ++r) {
        long ro = (long)(mTile * 16 + half * 8 + r) * N;
        Cmat[ro + col0     ] = acc0[r] + bi0;
        Cmat[ro + col0 + 16] = acc1[r] + bi1;
        Cmat[ro + col0 + 32] = acc2[r] + bi2;
        Cmat[ro + col0 + 48] = acc3[r] + bi3;
    }
}

// ---------------- per-step recurrent GEMM: z = xW[:,t,:] + h * Wh ----------------
// 64 blocks x 8 waves = 512 waves: 2 M-tiles (B=32) x 256 N-tiles (G=4096).
// All 8 waves in a block share the same 16 h-rows -> stage them in LDS once via
// async global->LDS copies (ASYNCcnt), then feed WMMA A-fragments from LDS.
constexpr int SH_STRIDE = H + 8;     // +16B row pad to spread LDS banks

__global__ void __launch_bounds__(256) lstm_step_gemm(
    const __bf16* __restrict__ hbf,  // [B,H] bf16
    const __bf16* __restrict__ WhT,  // [G,H] bf16 (Wh transposed)
    const float*  __restrict__ xW,   // [B*T, G] fp32, row = b*T + t
    float*        __restrict__ z,    // [B, G] fp32
    int t) {
    __shared__ __bf16 sh[16 * SH_STRIDE];          // ~33 KB

    int lane = threadIdx.x & 31;
    int waveInBlk = threadIdx.x >> 5;              // 0..7
    int mTile = blockIdx.x >> 5;                   // 0..1 (32 blocks per mTile)
    int nTile = ((blockIdx.x & 31) << 3) + waveInBlk;  // 0..255
    int l16 = lane & 15, half = lane >> 4;

    // ---- async stage h[mTile*16 .. +16, :] into LDS (16 rows x 2048 B) ----
    {
        unsigned ldsbase = (unsigned)(uintptr_t)(&sh[0]);
        const char* gsrc = (const char*)(hbf + (long)mTile * 16 * H);
        // 16 rows x 128 chunks of 16 B = 2048 chunks; 8 per thread
        for (int j = threadIdx.x; j < 2048; j += 256) {
            int row = j >> 7, chunk = j & 127;
            unsigned loff = ldsbase + (unsigned)(row * SH_STRIDE * 2 + chunk * 16);
            unsigned long long gaddr =
                (unsigned long long)(uintptr_t)(gsrc + row * (H * 2) + chunk * 16);
            asm volatile("global_load_async_to_lds_b128 %0, %1, off"
                         :: "v"(loff), "v"(gaddr)
                         : "memory");
        }
        asm volatile("s_wait_asynccnt 0" ::: "memory");
    }
    __syncthreads();

    const __bf16* arow = &sh[l16 * SH_STRIDE + half * 16];                 // LDS
    const __bf16* bcol = WhT + (long)(nTile * 16 + l16) * H + half * 16;   // L2

    v8f acc = {};
    for (int k0 = 0; k0 < H; k0 += 32) {
        v16bf a = *(const v16bf*)(arow + k0);      // ds_load_b128 x2
        v16bf b = *(const v16bf*)(bcol + k0);
        acc = __builtin_amdgcn_wmma_f32_16x16x32_bf16(false, a, false, b, (short)0, acc, false, false);
    }
    int col = nTile * 16 + l16;
#pragma unroll
    for (int r = 0; r < 8; ++r) {
        int brow = mTile * 16 + half * 8 + r;      // batch index 0..31
        z[(long)brow * G + col] =
            acc[r] + xW[((long)brow * T + t) * (long)G + col];
    }
}

// ---------------- per-step gates + state update ----------------
__global__ void lstm_step_gates(const float* __restrict__ z,  // [B,G]
                                float* __restrict__ c,        // [B,H]
                                __bf16* __restrict__ hbf,     // [B,H]
                                float* __restrict__ out,      // [B,T,H]
                                int t) {
    int i = blockIdx.x * blockDim.x + threadIdx.x;
    if (i >= B * H) return;
    int bi = i >> 10;        // /H
    int j  = i & (H - 1);
    const float* zb = z + (long)bi * G;
    float zi = zb[j], zf = zb[H + j], zg = zb[2 * H + j], zo = zb[3 * H + j];
    float si = 1.f / (1.f + __expf(-zi));
    float sf = 1.f / (1.f + __expf(-zf));
    float so = 1.f / (1.f + __expf(-zo));
    float cn = sf * c[i] + si * tanhf(zg);
    float hn = so * tanhf(cn);
    c[i]   = cn;
    hbf[i] = (__bf16)hn;
    out[((long)bi * T + t) * H + j] = hn;
}

__global__ void lstm_init(float* __restrict__ c, __bf16* __restrict__ hbf) {
    int i = blockIdx.x * blockDim.x + threadIdx.x;
    if (i < B * H) { c[i] = 0.f; hbf[i] = (__bf16)0.f; }
}

// ---------------- host orchestration ----------------
extern "C" void kernel_launch(void* const* d_in, const int* in_sizes, int n_in,
                              void* d_out, int out_size, void* d_ws, size_t ws_size,
                              hipStream_t stream) {
    const float* X    = (const float*)d_in[0]; // [B,T,D]
    const float* Wi   = (const float*)d_in[1]; // [D,G]
    const float* Wh   = (const float*)d_in[2]; // [H,G]
    const float* bias = (const float*)d_in[3]; // [G]
    float* out = (float*)d_out;                // [B,T,H]

    const long BT = (long)B * T;
    char* p = (char*)d_ws;
    __bf16* Xbf = (__bf16*)p; p += BT * D * sizeof(__bf16);        // 134 MB
    __bf16* WiT = (__bf16*)p; p += (long)G * D * sizeof(__bf16);   // 8 MB
    __bf16* WhT = (__bf16*)p; p += (long)G * H * sizeof(__bf16);   // 8 MB
    float*  xW  = (float*)p;  p += BT * (long)G * sizeof(float);   // 1.07 GB
    float*  z   = (float*)p;  p += (long)B * G * sizeof(float);
    float*  c   = (float*)p;  p += (long)B * H * sizeof(float);
    __bf16* hbf = (__bf16*)p; p += (long)B * H * sizeof(__bf16);

    // 1) bf16 conversions / weight transposes
    {
        long n = BT * D;
        cvt_bf16<<<(unsigned)((n + 255) / 256), 256, 0, stream>>>(X, Xbf, n);
    }
    {
        long n = (long)D * G;
        transpose_bf16<<<(unsigned)((n + 255) / 256), 256, 0, stream>>>(Wi, WiT, D, G);
    }
    {
        long n = (long)H * G;
        transpose_bf16<<<(unsigned)((n + 255) / 256), 256, 0, stream>>>(Wh, WhT, H, G);
    }

    // 2) Big input-projection GEMM: xW = Xbf * Wi + b
    {
        long waves   = (BT / 16) * (G / 64);     // 262144 waves
        long threads = waves * 32;
        gemm_bf16<<<(unsigned)(threads / 256), 256, 0, stream>>>(
            Xbf, WiT, bias, xW, (int)BT, D, G);
    }

    // 3) Recurrence
    lstm_init<<<(B * H + 255) / 256, 256, 0, stream>>>(c, hbf);
    for (int t = 0; t < T; ++t) {
        lstm_step_gemm<<<64, 256, 0, stream>>>(hbf, WhT, xW, z, t);
        lstm_step_gates<<<(B * H + 255) / 256, 256, 0, stream>>>(z, c, hbf, out, t);
    }
}
